// PyrConv3ChanSum_33337536152136
// MI455X (gfx1250) — matrix operand order repaired and checked by reading, not compiled, not run on hardware
//
#include <hip/hip_runtime.h>

typedef __attribute__((ext_vector_type(16))) _Float16 v16h;
typedef __attribute__((ext_vector_type(8)))  _Float16 v8h;
typedef __attribute__((ext_vector_type(8)))  float    v8f;

#define S_FULL 2048
#define NLEV   12

// ---------------------------------------------------------------- helpers
__device__ __forceinline__ float fetch0(const float* p, int s, int y, int x) {
  return (y >= 0 && y < s && x >= 0 && x < s) ? p[(size_t)y * s + x] : 0.0f;
}

__device__ __forceinline__ float bilin_clamp(const float* p, int s, float yf, float xf) {
  int y0 = (int)floorf(yf); float wy = yf - (float)y0;
  int x0 = (int)floorf(xf); float wx = xf - (float)x0;
  int y0c = min(max(y0, 0), s - 1);
  int y1c = min(max(y0 + 1, 0), s - 1);
  int x0c = min(max(x0, 0), s - 1);
  int x1c = min(max(x0 + 1, 0), s - 1);
  float a = p[(size_t)y0c * s + x0c];
  float b = p[(size_t)y0c * s + x1c];
  float c = p[(size_t)y1c * s + x0c];
  float d = p[(size_t)y1c * s + x1c];
  return (1.f - wy) * ((1.f - wx) * a + wx * b) + wy * ((1.f - wx) * c + wx * d);
}

// ---------------------------------------------------------------- gray (BGR planes)
__global__ void k_gray(const float* __restrict__ l, const float* __restrict__ r,
                       float* __restrict__ gl, float* __restrict__ gr) {
  const int HW = S_FULL * S_FULL;
  int idx = blockIdx.x * blockDim.x + threadIdx.x;
  if (idx >= HW) return;
  gl[idx] = 0.299f * l[2 * HW + idx] + 0.587f * l[HW + idx] + 0.114f * l[idx];
  gr[idx] = 0.299f * r[2 * HW + idx] + 0.587f * r[HW + idx] + 0.114f * r[idx];
}

// ---------------------------------------------------------------- 2x2 avgpool, 1ch f32
__global__ void k_down(const float* __restrict__ in, float* __restrict__ out, int sin) {
  int so = sin >> 1;
  int idx = blockIdx.x * blockDim.x + threadIdx.x;
  if (idx >= so * so) return;
  int y = idx / so, x = idx - y * so;
  const float* r0 = in + (size_t)(2 * y) * sin + 2 * x;
  out[idx] = 0.25f * (r0[0] + r0[1] + r0[sin] + r0[sin + 1]);
}

// ---------------------------------------------------------------- 2x2 avgpool, 24ch f16
__global__ void k_down24(const _Float16* __restrict__ in, _Float16* __restrict__ out, int sin) {
  int so = sin >> 1;
  size_t n = (size_t)24 * so * so;
  size_t idx = (size_t)blockIdx.x * blockDim.x + threadIdx.x;
  if (idx >= n) return;
  size_t pp = (size_t)so * so;
  int ch = (int)(idx / pp);
  int rem = (int)(idx - (size_t)ch * pp);
  int y = rem / so, x = rem - y * so;
  const _Float16* b = in + (size_t)ch * sin * sin + (size_t)(2 * y) * sin + 2 * x;
  float s0 = (float)b[0] + (float)b[1] + (float)b[sin] + (float)b[sin + 1];
  out[idx] = (_Float16)(0.25f * s0);
}

// ---------------------------------------------------------------- edge = |cur - up(next)| + eps, block-partial sumsq
__global__ void k_edge(const float* __restrict__ cur, const float* __restrict__ nxt,
                       float* __restrict__ outE, float* __restrict__ partial,
                       int s, int hasNext) {
  __shared__ float sm[256];
  int idx = blockIdx.x * blockDim.x + threadIdx.x;
  int n = s * s;
  float t = 0.f;
  if (idx < n) {
    int y = idx / s, x = idx - y * s;
    float e = cur[idx];
    if (hasNext) {
      int sn = s >> 1;
      e -= bilin_clamp(nxt, sn, (y + 0.5f) * 0.5f - 0.5f, (x + 0.5f) * 0.5f - 0.5f);
    }
    t = fabsf(e) + 1e-12f;
    outE[idx] = t;
  }
  sm[threadIdx.x] = t * t;
  __syncthreads();
  for (int o = 128; o > 0; o >>= 1) {
    if (threadIdx.x < o) sm[threadIdx.x] += sm[threadIdx.x + o];
    __syncthreads();
  }
  if (threadIdx.x == 0) partial[blockIdx.x] = sm[0];
}

// deterministic second-stage reduction (fixed tree, no atomics)
__global__ void k_reduce(const float* __restrict__ partial, int n, float* __restrict__ dst) {
  __shared__ float sm[256];
  float acc = 0.f;
  for (int i = threadIdx.x; i < n; i += 256) acc += partial[i];
  sm[threadIdx.x] = acc;
  __syncthreads();
  for (int o = 128; o > 0; o >>= 1) {
    if (threadIdx.x < o) sm[threadIdx.x] += sm[threadIdx.x + o];
    __syncthreads();
  }
  if (threadIdx.x == 0) dst[0] = sm[0];
}

// ---------------------------------------------------------------- pack W2 (24,2,3,3) into per-lane B fragments
// Layout: Bm[tile][lane][i] (tile<2, lane<32, i<16), contiguous 16 halves per lane
// Fragment semantics: lane holds column N = tile*16 + (lane&15), K = (lane>>4)*8 + i for i<8, 16 + (lane>>4)*8 + (i-8) for i>=8.
__global__ void k_prep_w2(const float* __restrict__ W2, _Float16* __restrict__ Bm) {
  int t = threadIdx.x;            // 0..1023 = tile*512 + lane*16 + i
  int tile = t >> 9;
  int lane = (t >> 4) & 31;
  int i    = t & 15;
  int k = (i < 8) ? (((lane >> 4) << 3) + i) : (16 + ((lane >> 4) << 3) + (i - 8));
  int n = (tile << 4) + (lane & 15);
  float v = 0.f;
  if (k < 18 && n < 24) {
    int c = (k >= 9);
    int tap = k - c * 9;
    v = W2[((n * 2 + c) * 3 + tap / 3) * 3 + (tap % 3)];
  }
  Bm[t] = (_Float16)v;
}

// ---------------------------------------------------------------- conv2: implicit GEMM via V_WMMA_F32_16X16X32_F16
// M = 16 output pixels (row strip), N = 24 channels (two 16-wide tiles), K = 18 (2ch x 3x3) padded to 32.
// Bias is folded into the WMMA C operand; full strips use packed b128 stores.
__global__ void k_conv2(const float* __restrict__ eL, const float* __restrict__ eR,
                        const float* __restrict__ norms, int lvl,
                        const _Float16* __restrict__ Bm, const float* __restrict__ b2,
                        _Float16* __restrict__ lc2, int s) {
  int wid = blockIdx.x * (blockDim.x >> 5) + (threadIdx.x >> 5);
  int spr = (s + 15) >> 4;
  int nstrips = s * spr;
  if (wid >= nstrips) return;                 // wave-uniform exit; WMMA waves keep EXEC all-ones

  int lane = threadIdx.x & 31;
  int mlo  = lane & 15;                       // A: M index; B/C: N index (within tile)
  int hi   = lane >> 4;

  int y  = wid / spr;
  int x0 = (wid - y * spr) << 4;
  int x  = x0 + mlo;

  float scaleL = rsqrtf(norms[lvl]);          // fold 1/frobenius-norm into A load
  float scaleR = rsqrtf(norms[12 + lvl]);

  __builtin_prefetch(eL + (size_t)y * s + x0, 0, 0);
  __builtin_prefetch(eR + (size_t)y * s + x0, 0, 0);

  // A fragment: lanes 0-15 hold K 0-7 & 16-23, lanes 16-31 hold K 8-15 & 24-31 (ISA 16-bit A 16x32 layout)
  int kA = hi * 8;
  v16h a;
#pragma unroll
  for (int i = 0; i < 16; ++i) {
    int k = (i < 8) ? (kA + i) : (16 + kA + (i - 8));
    float v = 0.0f;
    if (k < 18) {
      int c = (k >= 9);
      int tap = k - c * 9;
      int yy = y + tap / 3 - 1;
      int xx = x + tap % 3 - 1;
      v = c ? (fetch0(eR, s, yy, xx) * scaleR) : (fetch0(eL, s, yy, xx) * scaleL);
    }
    a[i] = (_Float16)v;
  }

  // B fragments: contiguous per-lane 32B vector loads (fragment-order packed by k_prep_w2)
  v16h b0 = *(const v16h*)(Bm + (size_t)lane * 16);
  v16h b1 = *(const v16h*)(Bm + 512 + (size_t)lane * 16);

  // Bias folded into C: lane's column is N = tile*16 + mlo for all 8 rows
  float bias0 = b2[mlo];
  float bias1 = (mlo < 8) ? b2[16 + mlo] : 0.f;
  v8f c0, c1;
#pragma unroll
  for (int i = 0; i < 8; ++i) { c0[i] = bias0; c1[i] = bias1; }

  c0 = __builtin_amdgcn_wmma_f32_16x16x32_f16(false, a, false, b0, (short)0, c0, false, false);
  c1 = __builtin_amdgcn_wmma_f32_16x16x32_f16(false, a, false, b1, (short)0, c1, false, false);

  // C/D layout: lane<16 -> N=lane, M=vgpr; lane>=16 -> N=lane-16, M=vgpr+8
  // Per lane the 8 rows are 8 consecutive x positions in one channel plane.
  size_t plane = (size_t)s * s;
  int mb = hi * 8;
  if (s >= 16) {
    // full strip: one aligned 16B store per tile (x0%16==0, mb%8==0, row stride %8==0)
    size_t rowoff = (size_t)y * s + x0 + mb;
    v8h h0;
#pragma unroll
    for (int i = 0; i < 8; ++i) h0[i] = (_Float16)c0[i];
    *(v8h*)(lc2 + (size_t)mlo * plane + rowoff) = h0;
    if (mlo < 8) {
      v8h h1;
#pragma unroll
      for (int i = 0; i < 8; ++i) h1[i] = (_Float16)c1[i];
      *(v8h*)(lc2 + (size_t)(16 + mlo) * plane + rowoff) = h1;
    }
  } else {
    // tiny levels: guarded scalar stores
#pragma unroll
    for (int vv = 0; vv < 8; ++vv) {
      int px = x0 + mb + vv;
      if (px < s) {
        lc2[(size_t)mlo * plane + (size_t)y * s + px] = (_Float16)c0[vv];
        if (mlo < 8)
          lc2[(size_t)(16 + mlo) * plane + (size_t)y * s + px] = (_Float16)c1[vv];
      }
    }
  }
}

// ---------------------------------------------------------------- conv3: 3x3 24->1, weights staged in LDS (N=1 => VALU dot)
__global__ void k_conv3(const _Float16* __restrict__ x, const float* __restrict__ W3,
                        const float* __restrict__ b3, float* __restrict__ out, int s) {
  __shared__ float wsh[216];
  __shared__ float bsh;
  if (threadIdx.x < 216) wsh[threadIdx.x] = W3[threadIdx.x];
  if (threadIdx.x == 0) bsh = b3[0];
  __syncthreads();
  int idx = blockIdx.x * blockDim.x + threadIdx.x;
  int n = s * s;
  if (idx >= n) return;
  int y = idx / s, xc = idx - y * s;
  size_t plane = (size_t)s * s;
  float acc = bsh;
  if (y > 0 && y < s - 1 && xc > 0 && xc < s - 1) {
    // interior fast path: no bounds checks
    const _Float16* p = x + (size_t)(y - 1) * s + (xc - 1);
    for (int c = 0; c < 24; ++c) {
      const _Float16* q = p + (size_t)c * plane;
      const float* wc = wsh + c * 9;
      __builtin_prefetch(q, 0, 0);
#pragma unroll
      for (int ky = 0; ky < 3; ++ky) {
        const _Float16* row = q + (size_t)ky * s;
        acc += (float)row[0] * wc[ky * 3 + 0] +
               (float)row[1] * wc[ky * 3 + 1] +
               (float)row[2] * wc[ky * 3 + 2];
      }
    }
  } else {
    for (int c = 0; c < 24; ++c) {
      const _Float16* p = x + (size_t)c * plane;
      const float* wc = wsh + c * 9;
#pragma unroll
      for (int ky = 0; ky < 3; ++ky) {
        int yy = y + ky - 1;
        if (yy < 0 || yy >= s) continue;
#pragma unroll
        for (int kx = 0; kx < 3; ++kx) {
          int xx = xc + kx - 1;
          if (xx < 0 || xx >= s) continue;
          acc += (float)p[(size_t)yy * s + xx] * wc[ky * 3 + kx];
        }
      }
    }
  }
  out[idx] = acc;
}

// ---------------------------------------------------------------- sum-depyramidalize inner pyramid: base + sum 0.5^j up(level j)
__global__ void k_depyr(const float* __restrict__ lc3, float* __restrict__ dst, int s0) {
  int idx = blockIdx.x * blockDim.x + threadIdx.x;
  int n = s0 * s0;
  if (idx >= n) return;
  int y = idx / s0, x = idx - y * s0;
  float acc = lc3[idx];
  size_t off = (size_t)s0 * s0;
  int t = s0 >> 1;
  float w = 0.5f;
  while (t >= 1) {
    float sc = (float)t / (float)s0;
    acc += w * bilin_clamp(lc3 + off, t, (y + 0.5f) * sc - 0.5f, (x + 0.5f) * sc - 0.5f);
    off += (size_t)t * t;
    t >>= 1;
    w *= 0.5f;
  }
  dst[idx] = acc;
}

// ---------------------------------------------------------------- 12-level upsample + softmax + expected depth + max-prob
__global__ void k_final(const float* __restrict__ d1, const float* __restrict__ bmul,
                        const float* __restrict__ badd, float* __restrict__ out) {
  const int S = S_FULL;
  const int HW = S * S;
  int idx = blockIdx.x * blockDim.x + threadIdx.x;
  if (idx >= HW) return;
  int y = idx / S, x = idx - y * S;
  float v[NLEV];
  size_t off = 0;
  int sk = S;
#pragma unroll
  for (int k = 0; k < NLEV; ++k) {
    if (sk == S) {
      v[k] = d1[idx];
    } else {
      float sc = (float)sk / (float)S;
      v[k] = bilin_clamp(d1 + off, sk, (y + 0.5f) * sc - 0.5f, (x + 0.5f) * sc - 0.5f);
    }
    off += (size_t)sk * sk;
    sk >>= 1;
  }
  float m = v[0];
#pragma unroll
  for (int k = 1; k < NLEV; ++k) m = fmaxf(m, v[k]);
  float sum = 0.f, dist = 0.f, bin = 1.f;
#pragma unroll
  for (int k = 0; k < NLEV; ++k) {
    float e = expf(v[k] - m);
    sum += e;
    dist += e * bin;
    bin *= 2.f;
  }
  out[idx] = (dist / sum) / bmul[0] + badd[0];
  out[HW + idx] = 1.f / sum;   // softmax max-prob = exp(0)/sum
}

// ---------------------------------------------------------------- host orchestration
extern "C" void kernel_launch(void* const* d_in, const int* in_sizes, int n_in,
                              void* d_out, int out_size, void* d_ws, size_t ws_size,
                              hipStream_t stream) {
  (void)in_sizes; (void)n_in; (void)out_size; (void)ws_size;
  const float* l    = (const float*)d_in[0];
  const float* r    = (const float*)d_in[1];
  const float* W2   = (const float*)d_in[2];
  const float* b2   = (const float*)d_in[3];
  const float* W3   = (const float*)d_in[4];
  const float* b3   = (const float*)d_in[5];
  const float* bmul = (const float*)d_in[6];
  const float* badd = (const float*)d_in[7];
  float* out = (float*)d_out;

  const int S = S_FULL;
  const size_t HW = (size_t)S * S;

  size_t OFF[NLEV + 1];
  OFF[0] = 0;
  for (int i = 0; i < NLEV; ++i) { int s = S >> i; OFF[i + 1] = OFF[i] + (size_t)s * s; }
  const size_t PYR = OFF[NLEV];   // 5,592,405 elements

  // bump allocator over d_ws (256B aligned regions)
  char* base = (char*)d_ws;
  size_t cur = 0;
  auto alloc = [&](size_t bytes) -> char* {
    char* p = base + cur;
    cur = (cur + bytes + 255) & ~(size_t)255;
    return p;
  };
  float*    gpL   = (float*)alloc(PYR * 4);
  float*    gpR   = (float*)alloc(PYR * 4);
  float*    eL    = (float*)alloc(PYR * 4);
  float*    eR    = (float*)alloc(PYR * 4);
  float*    norms = (float*)alloc(24 * 4);
  float*    part  = (float*)alloc(16384 * 4);
  float*    lc3   = (float*)alloc(PYR * 4);            // inner lc3 pyramid (reused per outer level)
  float*    dist1 = (float*)alloc(PYR * 4);
  _Float16* Bm    = (_Float16*)alloc(1024 * 2);
  _Float16* lc2   = (_Float16*)alloc((size_t)24 * PYR * 2); // inner 24ch pyramid (reused)

  auto GR = [](size_t n) -> unsigned { return (unsigned)((n + 255) / 256); };

  // 1) gray + gaussian pyramids
  k_gray<<<GR(HW), 256, 0, stream>>>(l, r, gpL, gpR);
  for (int i = 0; i < NLEV - 1; ++i) {
    int s = S >> i, so = s >> 1;
    k_down<<<GR((size_t)so * so), 256, 0, stream>>>(gpL + OFF[i], gpL + OFF[i + 1], s);
    k_down<<<GR((size_t)so * so), 256, 0, stream>>>(gpR + OFF[i], gpR + OFF[i + 1], s);
  }

  // 2) edge pyramids + deterministic per-level sumsq
  for (int i = 0; i < NLEV; ++i) {
    int s = S >> i;
    int hn = (i < NLEV - 1) ? 1 : 0;
    unsigned nb = GR((size_t)s * s);
    k_edge<<<nb, 256, 0, stream>>>(gpL + OFF[i], gpL + OFF[hn ? i + 1 : i], eL + OFF[i], part, s, hn);
    k_reduce<<<1, 256, 0, stream>>>(part, (int)nb, norms + i);
    k_edge<<<nb, 256, 0, stream>>>(gpR + OFF[i], gpR + OFF[hn ? i + 1 : i], eR + OFF[i], part, s, hn);
    k_reduce<<<1, 256, 0, stream>>>(part, (int)nb, norms + 12 + i);
  }

  // 3) pack conv2 weights into per-lane WMMA B fragments
  k_prep_w2<<<1, 1024, 0, stream>>>(W2, Bm);

  // 4) per outer level: WMMA conv2 -> 24ch inner pyramid -> conv3 -> depyramidalize
  for (int o = 0; o < NLEV; ++o) {
    int s = S >> o;
    int spr = (s + 15) >> 4;
    int nstrips = s * spr;
    k_conv2<<<(unsigned)((nstrips + 3) / 4), 128, 0, stream>>>(eL + OFF[o], eR + OFF[o],
                                                              norms, o, Bm, b2, lc2, s);
    {
      size_t io = 0; int t = s;
      while (t > 1) {
        int to = t >> 1;
        k_down24<<<GR((size_t)24 * to * to), 256, 0, stream>>>(lc2 + io, lc2 + io + (size_t)24 * t * t, t);
        io += (size_t)24 * t * t;
        t = to;
      }
    }
    {
      size_t io = 0, jo = 0; int t = s;
      for (;;) {
        k_conv3<<<GR((size_t)t * t), 256, 0, stream>>>(lc2 + io, W3, b3, lc3 + jo, t);
        io += (size_t)24 * t * t;
        jo += (size_t)t * t;
        if (t == 1) break;
        t >>= 1;
      }
    }
    k_depyr<<<GR((size_t)s * s), 256, 0, stream>>>(lc3, dist1 + OFF[o], s);
  }

  // 5) softmax over 12 depth bins -> (dist3, prob3)
  k_final<<<GR(HW), 256, 0, stream>>>(dist1, bmul, badd, out);
}
